// MOLERouter_87411174408786
// MI455X (gfx1250) — matrix operand order repaired and checked by reading, not compiled, not run on hardware
//
#include <hip/hip_runtime.h>
#include <hip/hip_bf16.h>

// Problem constants
#define IN_F   256
#define HID    128
#define NE     64
#define NROWS  1048576

#define WAVES           8
#define TILES_PER_WAVE  4
#define ROWS_PER_BLOCK  (WAVES * TILES_PER_WAVE * 16)   // 512
#define NBLOCKS         (NROWS / ROWS_PER_BLOCK)        // 2048

typedef float  v8f_t  __attribute__((ext_vector_type(8)));
typedef float  v4f_t  __attribute__((ext_vector_type(4)));
typedef __bf16 v16bf_t __attribute__((ext_vector_type(16)));
typedef __bf16 v8bf_t  __attribute__((ext_vector_type(8)));

__device__ __forceinline__ unsigned int pack_bf16(float a, float b) {
    union { __bf16 h[2]; unsigned int u; } x;
    x.h[0] = (__bf16)a;
    x.h[1] = (__bf16)b;
    return x.u;
}

// B-fragment load from pre-swizzled LDS: block = 1024B = 256 dwords,
// chunk0 (elements 0..7 of v16bf) at lane*16B, chunk1 at +512B.
__device__ __forceinline__ v16bf_t load_b_frag(const unsigned int* base, int blk, int lane) {
    const char* p = (const char*)(base + blk * 256 + lane * 4);
    v8bf_t lo = *(const v8bf_t*)p;
    v8bf_t hi = *(const v8bf_t*)(p + 512);
    return __builtin_shufflevector(lo, hi, 0,1,2,3,4,5,6,7,8,9,10,11,12,13,14,15);
}

__device__ __forceinline__ v8f_t wmma_bf16(v16bf_t a, v16bf_t b, v8f_t c) {
    // (neg_a, A, neg_b, B, c_mod, C, reuse_a, reuse_b)
    return __builtin_amdgcn_wmma_f32_16x16x32_bf16(false, a, false, b, (short)0, c, false, false);
}

__global__ __launch_bounds__(256, 1)
void MOLERouter_87411174408786_kernel(const float* __restrict__ X,
                                      const float* __restrict__ W1,
                                      const float* __restrict__ b1,
                                      const float* __restrict__ W2,
                                      const float* __restrict__ b2,
                                      float* __restrict__ out) {
    // LDS: W1 swizzled (64 blocks x 1KB), W2 swizzled (16 blocks x 1KB),
    // per-wave scratch 16 rows x 272B (h as bf16, then logits as f32)
    __shared__ unsigned int w1s[64 * 256];       // 65536 B
    __shared__ unsigned int w2s[16 * 256];       // 16384 B
    __shared__ unsigned int sws[WAVES * 16 * 68];// 34816 B

    const int tid  = threadIdx.x;
    const int wave = tid >> 5;
    const int lane = tid & 31;
    const int nn   = lane & 15;     // N (cols) / M (rows) sub-index
    const int hh   = lane >> 4;     // half selector
    const int h8   = hh * 8;

    // ---- Stage W1 into WMMA-B layout (bf16) ----
    // block = ct*8 + kt ; dword d in block: chunk=d>>7, l=(d>>2)&31, w=chunk*4+(d&3)
    for (int i = tid; i < 64 * 256; i += 256) {
        int blk   = i >> 8;
        int d     = i & 255;
        int chunk = d >> 7;
        int l     = (d >> 2) & 31;
        int w     = chunk * 4 + (d & 3);
        int ct    = blk >> 3;
        int kt    = blk & 7;
        int k     = kt * 32 + chunk * 16 + (l >> 4) * 8 + 2 * (d & 3);
        int c     = ct * 16 + (l & 15);
        w1s[i] = pack_bf16(W1[k * HID + c], W1[(k + 1) * HID + c]);
    }
    // ---- Stage W2 (block = ct*4 + kt, ct:0..3, kt:0..3) ----
    for (int i = tid; i < 16 * 256; i += 256) {
        int blk   = i >> 8;
        int d     = i & 255;
        int chunk = d >> 7;
        int l     = (d >> 2) & 31;
        int w     = chunk * 4 + (d & 3);
        int ct    = blk >> 2;
        int kt    = blk & 3;
        int k     = kt * 32 + chunk * 16 + (l >> 4) * 8 + 2 * (d & 3);
        int c     = ct * 16 + (l & 15);
        w2s[i] = pack_bf16(W2[k * NE + c], W2[(k + 1) * NE + c]);
    }
    __syncthreads();

    // Per-lane biases (col = tile*16 + nn for C-layout lanes)
    float b1v[8];
#pragma unroll
    for (int ct = 0; ct < 8; ++ct) b1v[ct] = b1[ct * 16 + nn];
    float b2v[4];
#pragma unroll
    for (int ct = 0; ct < 4; ++ct) b2v[ct] = b2[ct * 16 + nn];

    char* const swbase = (char*)sws + wave * 4352;   // 16 rows * 272B

#pragma unroll 1
    for (int t = 0; t < TILES_PER_WAVE; ++t) {
        const int row0 = blockIdx.x * ROWS_PER_BLOCK + (t * WAVES + wave) * 16;

        // ---- Load A fragments for GEMM1 (f32 -> bf16, WMMA A layout) ----
        const float* xr = X + (size_t)(row0 + nn) * IN_F;
        v16bf_t A[8];
#pragma unroll
        for (int ks = 0; ks < 8; ++ks) {
            const int k0 = ks * 32;
            v4f_t f0 = *(const v4f_t*)(xr + k0 + h8);
            v4f_t f1 = *(const v4f_t*)(xr + k0 + h8 + 4);
            v4f_t f2 = *(const v4f_t*)(xr + k0 + 16 + h8);
            v4f_t f3 = *(const v4f_t*)(xr + k0 + 16 + h8 + 4);
            v16bf_t a;
#pragma unroll
            for (int j = 0; j < 4; ++j) {
                a[j]      = (__bf16)f0[j];
                a[j + 4]  = (__bf16)f1[j];
                a[j + 8]  = (__bf16)f2[j];
                a[j + 12] = (__bf16)f3[j];
            }
            A[ks] = a;
        }

        // ---- GEMM1: h[16x128] = x[16x256] @ W1, 8 independent acc chains ----
        v8f_t acc1[8];
#pragma unroll
        for (int ct = 0; ct < 8; ++ct) acc1[ct] = (v8f_t){0,0,0,0,0,0,0,0};
#pragma unroll
        for (int ks = 0; ks < 8; ++ks) {
#pragma unroll
            for (int ct = 0; ct < 8; ++ct) {
                v16bf_t b = load_b_frag(w1s, ct * 8 + ks, lane);
                acc1[ct] = wmma_bf16(A[ks], b, acc1[ct]);
            }
        }

        // ---- bias + SiLU, store h as bf16 to LDS (row stride 136 bf16 = 272B) ----
        __bf16* hp = (__bf16*)swbase;
#pragma unroll
        for (int ct = 0; ct < 8; ++ct) {
#pragma unroll
            for (int r = 0; r < 8; ++r) {
                float v = acc1[ct][r] + b1v[ct];
                float s = v * __builtin_amdgcn_rcpf(1.0f + __expf(-v));
                hp[(r + 8 * hh) * 136 + ct * 16 + nn] = (__bf16)s;
            }
        }
        asm volatile("s_wait_dscnt 0x0" ::: "memory");

        // ---- Reload h in WMMA A layout (bf16) ----
        const char* hr = swbase + nn * 272;
        v16bf_t A2[4];
#pragma unroll
        for (int ks = 0; ks < 4; ++ks) {
            v8bf_t lo = *(const v8bf_t*)(hr + ks * 64 + hh * 16);
            v8bf_t hi = *(const v8bf_t*)(hr + ks * 64 + 32 + hh * 16);
            A2[ks] = __builtin_shufflevector(lo, hi, 0,1,2,3,4,5,6,7,8,9,10,11,12,13,14,15);
        }

        // ---- GEMM2: logits[16x64] = h @ W2 ----
        v8f_t acc2[4];
#pragma unroll
        for (int ct = 0; ct < 4; ++ct) acc2[ct] = (v8f_t){0,0,0,0,0,0,0,0};
#pragma unroll
        for (int ks = 0; ks < 4; ++ks) {
#pragma unroll
            for (int ct = 0; ct < 4; ++ct) {
                v16bf_t b = load_b_frag(w2s, ct * 4 + ks, lane);
                acc2[ct] = wmma_bf16(A2[ks], b, acc2[ct]);
            }
        }

        // ---- logits + b2 -> LDS f32 (row stride 68 dwords = 272B, reuse scratch) ----
        float* lg = (float*)swbase;
#pragma unroll
        for (int ct = 0; ct < 4; ++ct) {
#pragma unroll
            for (int r = 0; r < 8; ++r) {
                lg[(r + 8 * hh) * 68 + ct * 16 + nn] = acc2[ct][r] + b2v[ct];
            }
        }
        asm volatile("s_wait_dscnt 0x0" ::: "memory");

        // ---- top-8 mask + softmax: 2 lanes per row (row = nn, half = hh) ----
        {
            const float* lr = (const float*)(swbase + nn * 272) + hh * 32;
            float orig[32], work[32];
#pragma unroll
            for (int i = 0; i < 8; ++i)
                *(v4f_t*)(orig + 4 * i) = *(const v4f_t*)(lr + 4 * i);
#pragma unroll
            for (int i = 0; i < 32; ++i) work[i] = orig[i];

            const float NEGINF = -__builtin_inff();
            float thresh = 0.0f;   // 8th-largest after loop
            float gmax   = 0.0f;   // largest
#pragma unroll 1
            for (int it = 0; it < 8; ++it) {
                float cur = NEGINF;
#pragma unroll
                for (int i = 0; i < 32; ++i) cur = fmaxf(cur, work[i]);
                float oth = __shfl_xor(cur, 16, 32);
                float g = fmaxf(cur, oth);
                if (it == 0) gmax = g;
#pragma unroll
                for (int i = 0; i < 32; ++i)
                    work[i] = (work[i] == g) ? NEGINF : work[i];
                thresh = g;
            }

            float sum = 0.0f;
#pragma unroll
            for (int i = 0; i < 32; ++i) {
                float e = __expf(orig[i] - gmax);
                float m = (orig[i] >= thresh) ? e : 0.0f;
                orig[i] = m;
                sum += m;
            }
            sum += __shfl_xor(sum, 16, 32);
            const float rs = __builtin_amdgcn_rcpf(sum);

            float* op = out + (size_t)(row0 + nn) * NE + hh * 32;
#pragma unroll
            for (int i = 0; i < 8; ++i) {
                v4f_t w;
#pragma unroll
                for (int j = 0; j < 4; ++j) w[j] = orig[4 * i + j] * rs;
                *(v4f_t*)(op + 4 * i) = w;
            }
        }
    }
}

extern "C" void kernel_launch(void* const* d_in, const int* in_sizes, int n_in,
                              void* d_out, int out_size, void* d_ws, size_t ws_size,
                              hipStream_t stream) {
    (void)in_sizes; (void)n_in; (void)d_ws; (void)ws_size; (void)out_size;
    const float* X  = (const float*)d_in[0];
    const float* W1 = (const float*)d_in[1];
    const float* b1 = (const float*)d_in[2];
    const float* W2 = (const float*)d_in[3];
    const float* b2 = (const float*)d_in[4];
    float* out = (float*)d_out;

    MOLERouter_87411174408786_kernel<<<NBLOCKS, 256, 0, stream>>>(X, W1, b1, W2, b2, out);
}
